// Attention_3040836845914
// MI455X (gfx1250) — matrix operand order repaired and checked by reading, not compiled, not run on hardware
//
#include <hip/hip_runtime.h>

// MHA forward for MI455X (gfx1250, wave32, WMMA f16 16x16x32, f32 accum).
// Workspace requirement: ~48 MB (Xh 8MB, 4 transposed f16 weights 8MB,
// maskbias 16KB, Q/K/Vt 24MB, ctx 8MB).

typedef __attribute__((ext_vector_type(16))) _Float16 v16h;
typedef __attribute__((ext_vector_type(8)))  float    v8f;

#define HD_   64
#define NH_   16
#define HID_  1024
#define B_    2
#define S_    2048
#define MROWS (B_ * S_)   // 4096 token rows

union AFrag { v16h v; uint4 q[2]; };
union HPack { _Float16 h[8]; uint4 u; };

static __device__ __forceinline__ v8f wmma16(v16h a, v16h b, v8f c) {
  // D = A(16x32 f16) * B(32x16 f16) + C(16x16 f32)
  return __builtin_amdgcn_wmma_f32_16x16x32_f16(false, a, false, b, (short)0, c,
                                                false, false);
}

// ---------------------------------------------------------------- converts --
__global__ void f32_to_f16_vec(const float* __restrict__ in,
                               _Float16* __restrict__ out, int n4) {
  int i = blockIdx.x * blockDim.x + threadIdx.x;
  if (i >= n4) return;
  float4 v = reinterpret_cast<const float4*>(in)[i];
  union { _Float16 h[4]; uint2 u; } p;
  p.h[0] = (_Float16)v.x; p.h[1] = (_Float16)v.y;
  p.h[2] = (_Float16)v.z; p.h[3] = (_Float16)v.w;
  reinterpret_cast<uint2*>(out)[i] = p.u;
}

// W is [K][N] row-major f32; produce Wt [N][K] f16 (so B-fragments are
// contiguous 32B loads along K).
__global__ void transpose_to_f16(const float* __restrict__ W,
                                 _Float16* __restrict__ Wt, int K, int N) {
  int i = blockIdx.x * blockDim.x + threadIdx.x;
  if (i >= N * K) return;
  int n = i / K;
  int k = i - n * K;
  Wt[i] = (_Float16)W[(size_t)k * N + n];
}

__global__ void mask_to_bias(const float* __restrict__ mask,
                             float* __restrict__ mb, int n) {
  int i = blockIdx.x * blockDim.x + threadIdx.x;
  if (i < n) mb[i] = -1.0e6f * (1.0f - mask[i]);
}

// ---------------------------------------------------------------- QKV GEMM --
// out = Xh[M,1024] @ Wt^T (+bias), one wave -> 16(M) x 64(N) tile.
// mode 0: out[(b*NH+head)*S + s][d]  (row-major heads, scaled)
// mode 2: out[(b*NH+head)*HD + d][s] (V transposed for PV B-fragments)
__global__ __launch_bounds__(256) void gemm_qkv(
    const _Float16* __restrict__ Xh, const _Float16* __restrict__ Wt,
    const float* __restrict__ bias, _Float16* __restrict__ out,
    int mode, float scale) {
  const int lane = threadIdx.x & 31;
  const int wave = threadIdx.x >> 5;
  const int lo = lane & 15, hi = lane >> 4;
  const int m0 = blockIdx.x * 16;
  const int n0 = (blockIdx.y * 8 + wave) * 64;

  v8f zv = {};
  v8f acc[4] = {zv, zv, zv, zv};

  const _Float16* arow = Xh + (size_t)(m0 + lo) * HID_;
  for (int k0 = 0; k0 < HID_; k0 += 32) {
    AFrag a;
    a.q[0] = *reinterpret_cast<const uint4*>(arow + k0 + hi * 16);
    a.q[1] = *reinterpret_cast<const uint4*>(arow + k0 + hi * 16 + 8);
#pragma unroll
    for (int j = 0; j < 4; ++j) {
      const _Float16* brow =
          Wt + (size_t)(n0 + j * 16 + lo) * HID_ + k0 + hi * 16;
      v16h b = *reinterpret_cast<const v16h*>(brow);
      acc[j] = wmma16(a.v, b, acc[j]);
    }
  }

#pragma unroll
  for (int j = 0; j < 4; ++j) {
    const int n = n0 + j * 16 + lo;
    const float bv = bias[n];
    const int head = n >> 6, d = n & 63;
#pragma unroll
    for (int r = 0; r < 8; ++r) {
      const int sg = m0 + r + hi * 8;          // global token row
      const int b = sg >> 11, s = sg & 2047;
      const float val = (acc[j][r] + bv) * scale;
      size_t off;
      if (mode == 2)
        off = ((size_t)((b * NH_ + head) * HD_ + d)) * S_ + s;
      else
        off = ((size_t)((b * NH_ + head) * S_ + s)) * HD_ + d;
      out[off] = (_Float16)val;
    }
  }
}

// --------------------------------------------------------------- attention --
// One wave per (b, head, 16-query tile). Computes S^T = K*Q^T so softmax
// stats are per-lane (query = lane%16); P^T -> A-layout via padded LDS.
__global__ __launch_bounds__(32) void attn_kernel(
    const _Float16* __restrict__ Qh, const _Float16* __restrict__ Kh,
    const _Float16* __restrict__ Vt, const float* __restrict__ maskb,
    _Float16* __restrict__ ctx) {
  __shared__ __align__(16) _Float16 Plds[16 * 40];  // 16 q-rows, pitch 40
  __shared__ __align__(16) float alpha_l[16];
  __shared__ __align__(16) float linv_l[16];

  const int lane = threadIdx.x & 31;
  const int lo = lane & 15, hi = lane >> 4;
  const int idx = blockIdx.x;
  const int qt = idx & 127;
  const int h = (idx >> 7) & (NH_ - 1);
  const int b = idx >> 11;
  const int q0 = qt * 16;

  const _Float16* Qp = Qh + (size_t)(b * NH_ + h) * S_ * HD_;
  const _Float16* Kp = Kh + (size_t)(b * NH_ + h) * S_ * HD_;
  const _Float16* Vp = Vt + (size_t)(b * NH_ + h) * HD_ * S_;
  const float* mb = maskb + (size_t)b * S_;

  // B-fragments of Q^T (d = 32*j + 16*hi + e), resident for whole loop.
  const v16h bq0 =
      *reinterpret_cast<const v16h*>(Qp + (size_t)(q0 + lo) * HD_ + hi * 16);
  const v16h bq1 = *reinterpret_cast<const v16h*>(
      Qp + (size_t)(q0 + lo) * HD_ + 32 + hi * 16);

  v8f zv = {};
  v8f acc[4] = {zv, zv, zv, zv};
  float mrun = -3.0e38f, lrun = 0.0f;

  for (int kb = 0; kb < S_; kb += 32) {
    v8f st[2];
#pragma unroll
    for (int t = 0; t < 2; ++t) {
      const int kt = kb + t * 16;
      const _Float16* krow = Kp + (size_t)(kt + lo) * HD_;
      AFrag a0, a1;
      a0.q[0] = *reinterpret_cast<const uint4*>(krow + hi * 16);
      a0.q[1] = *reinterpret_cast<const uint4*>(krow + hi * 16 + 8);
      a1.q[0] = *reinterpret_cast<const uint4*>(krow + 32 + hi * 16);
      a1.q[1] = *reinterpret_cast<const uint4*>(krow + 40 + hi * 16);
      v8f s = zv;
      s = wmma16(a0.v, bq0, s);
      s = wmma16(a1.v, bq1, s);
      // additive mask bias per key row (keys kt + r + 8*hi)
      const float4 ma = *reinterpret_cast<const float4*>(mb + kt + hi * 8);
      const float4 mc = *reinterpret_cast<const float4*>(mb + kt + hi * 8 + 4);
      s[0] += ma.x; s[1] += ma.y; s[2] += ma.z; s[3] += ma.w;
      s[4] += mc.x; s[5] += mc.y; s[6] += mc.z; s[7] += mc.w;
      st[t] = s;
    }

    // online softmax (per query = lane%16; halves hold disjoint key sets)
    float mx = st[0][0];
#pragma unroll
    for (int t = 0; t < 2; ++t)
#pragma unroll
      for (int r = 0; r < 8; ++r) mx = fmaxf(mx, st[t][r]);
    mx = fmaxf(mx, __shfl_xor(mx, 16, 32));
    const float mnew = fmaxf(mrun, mx);
    const float alpha = __expf(mrun - mnew);
    float rs = 0.0f;
#pragma unroll
    for (int t = 0; t < 2; ++t)
#pragma unroll
      for (int r = 0; r < 8; ++r) {
        const float p = __expf(st[t][r] - mnew);
        st[t][r] = p;
        rs += p;
      }
    rs += __shfl_xor(rs, 16, 32);
    lrun = lrun * alpha + rs;
    mrun = mnew;

    // P^T tiles -> LDS as P[q][key_in_block] (conflict-free 16B stores)
#pragma unroll
    for (int t = 0; t < 2; ++t) {
      HPack pk;
#pragma unroll
      for (int r = 0; r < 8; ++r) pk.h[r] = (_Float16)st[t][r];
      *reinterpret_cast<uint4*>(&Plds[lo * 40 + t * 16 + hi * 8]) = pk.u;
    }
    alpha_l[lo] = alpha;  // duplicated write from both halves, same value
    __syncthreads();      // single-wave WG: just orders LDS (dscnt wait)

    // rescale running O by alpha[query]
    const float4 af0 = *reinterpret_cast<const float4*>(&alpha_l[hi * 8]);
    const float4 af1 = *reinterpret_cast<const float4*>(&alpha_l[hi * 8 + 4]);
    const float af[8] = {af0.x, af0.y, af0.z, af0.w, af1.x, af1.y, af1.z, af1.w};
#pragma unroll
    for (int j = 0; j < 4; ++j)
#pragma unroll
      for (int r = 0; r < 8; ++r) acc[j][r] *= af[r];

    // A = P (16q x 32key) from LDS in A-layout
    AFrag ap;
    ap.q[0] = *reinterpret_cast<const uint4*>(&Plds[lo * 40 + hi * 16]);
    ap.q[1] = *reinterpret_cast<const uint4*>(&Plds[lo * 40 + hi * 16 + 8]);
#pragma unroll
    for (int j = 0; j < 4; ++j) {
      const v16h bv = *reinterpret_cast<const v16h*>(
          Vp + (size_t)(j * 16 + lo) * S_ + kb + hi * 16);
      acc[j] = wmma16(ap.v, bv, acc[j]);
    }
    __syncthreads();
  }

  linv_l[lo] = 1.0f / lrun;
  __syncthreads();
  const float4 lf0 = *reinterpret_cast<const float4*>(&linv_l[hi * 8]);
  const float4 lf1 = *reinterpret_cast<const float4*>(&linv_l[hi * 8 + 4]);
  const float lf[8] = {lf0.x, lf0.y, lf0.z, lf0.w, lf1.x, lf1.y, lf1.z, lf1.w};

#pragma unroll
  for (int j = 0; j < 4; ++j)
#pragma unroll
    for (int r = 0; r < 8; ++r) {
      const size_t row = (size_t)b * S_ + q0 + r + hi * 8;
      ctx[row * (NH_ * HD_) + h * HD_ + j * 16 + lo] =
          (_Float16)(acc[j][r] * lf[r]);
    }
}

// ----------------------------------------------------------- output GEMM ---
__global__ __launch_bounds__(256) void gemm_out_f32(
    const _Float16* __restrict__ ctx, const _Float16* __restrict__ Wot,
    const float* __restrict__ bo, float* __restrict__ out) {
  const int lane = threadIdx.x & 31;
  const int wave = threadIdx.x >> 5;
  const int lo = lane & 15, hi = lane >> 4;
  const int m0 = blockIdx.x * 16;
  const int n0 = (blockIdx.y * 8 + wave) * 64;

  v8f zv = {};
  v8f acc[4] = {zv, zv, zv, zv};

  const _Float16* arow = ctx + (size_t)(m0 + lo) * HID_;
  for (int k0 = 0; k0 < HID_; k0 += 32) {
    AFrag a;
    a.q[0] = *reinterpret_cast<const uint4*>(arow + k0 + hi * 16);
    a.q[1] = *reinterpret_cast<const uint4*>(arow + k0 + hi * 16 + 8);
#pragma unroll
    for (int j = 0; j < 4; ++j) {
      const _Float16* brow =
          Wot + (size_t)(n0 + j * 16 + lo) * HID_ + k0 + hi * 16;
      v16h bmat = *reinterpret_cast<const v16h*>(brow);
      acc[j] = wmma16(a.v, bmat, acc[j]);
    }
  }

#pragma unroll
  for (int j = 0; j < 4; ++j) {
    const int n = n0 + j * 16 + lo;
    const float bv = bo[n];
#pragma unroll
    for (int r = 0; r < 8; ++r) {
      const size_t sg = m0 + r + hi * 8;
      out[sg * HID_ + n] = acc[j][r] + bv;
    }
  }
}

// ------------------------------------------------------------------ launch --
extern "C" void kernel_launch(void* const* d_in, const int* in_sizes, int n_in,
                              void* d_out, int out_size, void* d_ws,
                              size_t ws_size, hipStream_t stream) {
  const float* X  = (const float*)d_in[0];
  const float* mask = (const float*)d_in[1];
  const float* Wq = (const float*)d_in[2];
  const float* bq = (const float*)d_in[3];
  const float* Wk = (const float*)d_in[4];
  const float* bk = (const float*)d_in[5];
  const float* Wv = (const float*)d_in[6];
  const float* bv = (const float*)d_in[7];
  const float* Wo = (const float*)d_in[8];
  const float* bo = (const float*)d_in[9];
  float* out = (float*)d_out;

  char* ws = (char*)d_ws;
  _Float16* Xh  = (_Float16*)ws; ws += (size_t)MROWS * HID_ * 2;      // 8 MB
  _Float16* Wqt = (_Float16*)ws; ws += (size_t)HID_ * HID_ * 2;      // 2 MB
  _Float16* Wkt = (_Float16*)ws; ws += (size_t)HID_ * HID_ * 2;
  _Float16* Wvt = (_Float16*)ws; ws += (size_t)HID_ * HID_ * 2;
  _Float16* Wot = (_Float16*)ws; ws += (size_t)HID_ * HID_ * 2;
  float*    mbb = (float*)ws;    ws += (size_t)B_ * S_ * 4;          // 16 KB
  _Float16* Qh  = (_Float16*)ws; ws += (size_t)MROWS * HID_ * 2;     // 8 MB
  _Float16* Kh  = (_Float16*)ws; ws += (size_t)MROWS * HID_ * 2;
  _Float16* Vth = (_Float16*)ws; ws += (size_t)MROWS * HID_ * 2;
  _Float16* ctx = (_Float16*)ws; ws += (size_t)MROWS * HID_ * 2;

  const int n4 = MROWS * HID_ / 4;
  f32_to_f16_vec<<<(n4 + 255) / 256, 256, 0, stream>>>(X, Xh, n4);
  const int nw = HID_ * HID_;
  transpose_to_f16<<<(nw + 255) / 256, 256, 0, stream>>>(Wq, Wqt, HID_, HID_);
  transpose_to_f16<<<(nw + 255) / 256, 256, 0, stream>>>(Wk, Wkt, HID_, HID_);
  transpose_to_f16<<<(nw + 255) / 256, 256, 0, stream>>>(Wv, Wvt, HID_, HID_);
  transpose_to_f16<<<(nw + 255) / 256, 256, 0, stream>>>(Wo, Wot, HID_, HID_);
  mask_to_bias<<<(B_ * S_ + 255) / 256, 256, 0, stream>>>(mask, mbb, B_ * S_);

  dim3 ggrid(MROWS / 16, 2);  // 256 M-tiles x (16 N64-tiles / 8 waves)
  gemm_qkv<<<ggrid, 256, 0, stream>>>(Xh, Wqt, bq, Qh, 0, 0.125f);  // Q*1/sqrt(64)
  gemm_qkv<<<ggrid, 256, 0, stream>>>(Xh, Wkt, bk, Kh, 0, 1.0f);
  gemm_qkv<<<ggrid, 256, 0, stream>>>(Xh, Wvt, bv, Vth, 2, 1.0f);   // V transposed

  attn_kernel<<<B_ * NH_ * (S_ / 16), 32, 0, stream>>>(Qh, Kh, Vth, mbb, ctx);

  gemm_out_f32<<<ggrid, 256, 0, stream>>>(ctx, Wot, bo, out);
}
